// SequentialGOPTracker_4818953306572
// MI455X (gfx1250) — matrix operand order repaired and checked by reading, not compiled, not run on hardware
//
#include <hip/hip_runtime.h>
#include <hip/hip_bf16.h>
#include <stdint.h>

#define B_   128
#define M_   100
#define H_   128
#define BM   12800          // B_*M_
#define KIN  192            // H_ + 64
#define G4   512            // 4*H_

typedef __attribute__((ext_vector_type(16))) __bf16 v16bf;
typedef __attribute__((ext_vector_type(8)))  float  v8f;

union FragBF {
  v16bf v;
  uint4 q[2];
};

__device__ __forceinline__ uint16_t f32_to_bf16_rte(float f) {
  uint32_t u = __float_as_uint(f);
  u += 0x7FFFu + ((u >> 16) & 1u);
  return (uint16_t)(u >> 16);
}
__device__ __forceinline__ float bf16lo(uint32_t p) { return __uint_as_float(p << 16); }
__device__ __forceinline__ float bf16hi(uint32_t p) { return __uint_as_float(p & 0xFFFF0000u); }

__device__ __forceinline__ float fsigmoid(float x) { return 1.0f / (1.0f + __expf(-x)); }
__device__ __forceinline__ float ftanh(float x)    { return 1.0f - 2.0f / (__expf(2.0f * x) + 1.0f); }

// ---------------------------------------------------------------- f32 -> bf16
__global__ void cvt_f32_bf16(const float* __restrict__ in, uint16_t* __restrict__ o, int n) {
  int i = blockIdx.x * 256 + threadIdx.x;
  if (i < n) o[i] = f32_to_bf16_rte(in[i]);
}

// ---------------------------------------------------------------- conv1 + relu
// out [128,32,40,40] f32
__global__ __launch_bounds__(256) void conv1_kernel(
    const float* __restrict__ mv, const float* __restrict__ w,
    const float* __restrict__ bias, float* __restrict__ o) {
  int idx = blockIdx.x * 256 + threadIdx.x;
  if (idx >= B_ * 32 * 1600) return;
  int x  = idx % 40;
  int y  = (idx / 40) % 40;
  int oc = (idx / 1600) % 32;
  int bn = idx / (1600 * 32);
  float s = bias[oc];
  #pragma unroll
  for (int ic = 0; ic < 2; ++ic) {
    const float* base = mv + ((size_t)bn * 2 + ic) * 1600;
    #pragma unroll
    for (int ky = 0; ky < 3; ++ky) {
      int iy = y + ky - 1;
      if (iy < 0 || iy > 39) continue;
      #pragma unroll
      for (int kx = 0; kx < 3; ++kx) {
        int ix = x + kx - 1;
        if (ix < 0 || ix > 39) continue;
        s = fmaf(base[iy * 40 + ix], w[((oc * 2 + ic) * 3 + ky) * 3 + kx], s);
      }
    }
  }
  o[idx] = fmaxf(s, 0.0f);
}

// ------------------------------------------- conv2 + relu + 8x8 avg-pool fused
// 8 output channels per block: each loaded input tap feeds 8 FMAs (8x less L2
// traffic than one-oc-per-block). pooled [128,4096] bf16 (oc*64 + py*8 + px).
__global__ __launch_bounds__(256) void conv2_pool_kernel(
    const float* __restrict__ c1, const float* __restrict__ w,
    const float* __restrict__ bias, uint16_t* __restrict__ pooled) {
  const int bn  = blockIdx.x >> 3;        // 128 batches
  const int ocg = (blockIdx.x & 7) * 8;   // base of 8-channel group
  __shared__ float bins[8][64];
  __shared__ float wl[8][288];
  for (int i = threadIdx.x; i < 8 * 64; i += 256) bins[i >> 6][i & 63] = 0.0f;
  for (int i = threadIdx.x; i < 8 * 288; i += 256)
    wl[i / 288][i % 288] = w[(size_t)(ocg + i / 288) * 288 + (i % 288)];
  float bacc[8];
  #pragma unroll
  for (int o = 0; o < 8; ++o) bacc[o] = bias[ocg + o];
  __syncthreads();
  for (int p = threadIdx.x; p < 1600; p += 256) {
    int x = p % 40, y = p / 40;
    float acc[8];
    #pragma unroll
    for (int o = 0; o < 8; ++o) acc[o] = bacc[o];
    for (int ic = 0; ic < 32; ++ic) {
      const float* base = c1 + ((size_t)bn * 32 + ic) * 1600;
      #pragma unroll
      for (int ky = 0; ky < 3; ++ky) {
        int iy = y + ky - 1;
        if (iy < 0 || iy > 39) continue;
        #pragma unroll
        for (int kx = 0; kx < 3; ++kx) {
          int ix = x + kx - 1;
          if (ix < 0 || ix > 39) continue;
          const float v = base[iy * 40 + ix];
          const int widx = ic * 9 + ky * 3 + kx;
          #pragma unroll
          for (int o = 0; o < 8; ++o) acc[o] = fmaf(v, wl[o][widx], acc[o]);
        }
      }
    }
    const int bin = (y / 5) * 8 + (x / 5);
    #pragma unroll
    for (int o = 0; o < 8; ++o)
      atomicAdd(&bins[o][bin], fmaxf(acc[o], 0.0f) * (1.0f / 25.0f));  // ds_add_f32
  }
  __syncthreads();
  for (int i = threadIdx.x; i < 8 * 64; i += 256) {
    const int o = i >> 6, bin = i & 63;
    pooled[(size_t)bn * 4096 + (size_t)(ocg + o) * 64 + bin] = f32_to_bf16_rte(bins[o][bin]);
  }
}

// --------------------------------------------------------- WMMA GEMM C = A*W^T
// A [M,K] bf16 row-major, W [N,K] bf16 row-major, C [M,N] f32, +optional biases.
// block = 256 threads (8 waves); wave = one 16x16 D tile.
__global__ __launch_bounds__(256) void wmma_gemm_xwT(
    const uint16_t* __restrict__ A, const uint16_t* __restrict__ W,
    const float* __restrict__ bias0, const float* __restrict__ bias1,
    float* __restrict__ C, int M, int N, int K) {
  const int lane = threadIdx.x & 31;
  const int wave = threadIdx.x >> 5;
  const int half = lane >> 4;
  const int lm   = lane & 15;
  const int m0 = blockIdx.x * 16;
  const int n0 = (blockIdx.y * 8 + wave) * 16;
  if (m0 >= M || n0 >= N) return;  // wave-uniform
  const uint16_t* arow = A + (size_t)(m0 + lm) * K;
  const uint16_t* wrow = W + (size_t)(n0 + lm) * K;
  v8f acc = {};
  for (int k0 = 0; k0 < K; k0 += 32) {
    FragBF fa, fb;
    fa.q[0] = *(const uint4*)(arow + k0 + half * 8);        // K = k0+half*8 .. +7
    fa.q[1] = *(const uint4*)(arow + k0 + 16 + half * 8);   // K = k0+16+half*8 .. +7
    const uint4* pb = (const uint4*)(wrow + k0 + half * 8);
    fb.q[0] = pb[0];
    fb.q[1] = pb[2];                                        // K = k0+half*8 .. +7, k0+16+half*8 .. +7
    acc = __builtin_amdgcn_wmma_f32_16x16x32_bf16(false, fa.v, false, fb.v,
                                                  (short)0, acc, false, false);
  }
  float bv = 0.0f;
  if (bias0) bv += bias0[n0 + lm];
  if (bias1) bv += bias1[n0 + lm];
  #pragma unroll
  for (int r = 0; r < 8; ++r)
    C[(size_t)(m0 + half * 8 + r) * N + (n0 + lm)] = acc[r] + bv;
}

// ----------------------------------- combined[12800,192] bf16 (motion | boxfeat)
__global__ void build_combined_kernel(
    const float* __restrict__ motion_feat, const float* __restrict__ prev_boxes,
    const float* __restrict__ bbox_w, const float* __restrict__ bbox_b,
    uint16_t* __restrict__ comb) {
  int idx = blockIdx.x * 256 + threadIdx.x;
  if (idx >= BM * KIN) return;
  int col = idx % KIN;
  int row = idx / KIN;
  int bn  = row / M_;
  float v;
  if (col < H_) {
    v = motion_feat[(size_t)bn * H_ + col];
  } else {
    int j = col - H_;
    const float* pb = prev_boxes + (size_t)row * 4;
    const float* wr = bbox_w + (size_t)j * 4;
    v = bbox_b[j] + pb[0] * wr[0] + pb[1] * wr[1] + pb[2] * wr[2] + pb[3] * wr[3];
  }
  comb[idx] = f32_to_bf16_rte(v);
}

// ------------------------------------------------ sequential 2-layer LSTM scan
// One persistent 512-thread workgroup. Recurrent matrices live in VGPRs
// (64 packed-bf16 dwords per matrix row per thread); h-state/gates in LDS,
// c-state in registers of the owning thread. X0 row software-pipelined one
// step ahead in registers + global_prefetch two steps ahead. 3 barriers/step.
__global__ __launch_bounds__(512, 1) void lstm_scan_kernel(
    const float* __restrict__ X0,        // [BM,512] precomputed x@W_ih0^T + b
    const uint16_t* __restrict__ Whh0,   // [512,128] bf16
    const uint16_t* __restrict__ Wih1,   // [512,128] bf16
    const uint16_t* __restrict__ Whh1,   // [512,128] bf16
    const float* __restrict__ b_ih1, const float* __restrict__ b_hh1,
    uint16_t* __restrict__ outs) {       // [BM,128] bf16
  __shared__ float h0s[H_], h1s[H_];
  __shared__ float g0s[G4], g1s[G4];
  const int g = threadIdx.x;

  uint32_t w0[64], w1[64], w2[64];
  {
    const uint4* p0 = (const uint4*)(Whh0 + (size_t)g * H_);
    const uint4* p1 = (const uint4*)(Wih1 + (size_t)g * H_);
    const uint4* p2 = (const uint4*)(Whh1 + (size_t)g * H_);
    #pragma unroll
    for (int i = 0; i < 16; ++i) {
      uint4 a = p0[i]; w0[4*i]=a.x; w0[4*i+1]=a.y; w0[4*i+2]=a.z; w0[4*i+3]=a.w;
      uint4 b = p1[i]; w1[4*i]=b.x; w1[4*i+1]=b.y; w1[4*i+2]=b.z; w1[4*i+3]=b.w;
      uint4 c = p2[i]; w2[4*i]=c.x; w2[4*i+1]=c.y; w2[4*i+2]=c.z; w2[4*i+3]=c.w;
    }
  }
  const float bias1 = b_ih1[g] + b_hh1[g];
  const int gate = g >> 7;      // 0=i 1=f 2=g 3=o (torch gate order)
  const int j = g & (H_ - 1);
  float c0r = 0.0f, c1r = 0.0f;            // private cell state (thread j owns)
  if (g < H_) { h0s[g] = 0.0f; h1s[g] = 0.0f; }
  float xcur = X0[g];                      // pipelined input projection
  __syncthreads();

  for (int t = 0; t < BM; ++t) {
    // ---- phase 1: layer-0 gate = hoisted input proj + Whh0*h0
    float a = xcur;
    {
      const int tn = (t + 1 < BM) ? (t + 1) : (BM - 1);
      xcur = X0[(size_t)tn * G4 + g];                       // hide L2 latency
      const int tp = (t + 2 < BM) ? (t + 2) : (BM - 1);
      __builtin_prefetch(X0 + (size_t)tp * G4 + g, 0, 3);   // global_prefetch_b8
    }
    #pragma unroll
    for (int i = 0; i < 64; ++i) {
      const uint32_t p = w0[i];
      a = fmaf(bf16lo(p), h0s[2*i],     a);
      a = fmaf(bf16hi(p), h0s[2*i + 1], a);
    }
    g0s[g] = (gate == 2) ? ftanh(a) : fsigmoid(a);
    __syncthreads();                                        // B1: g0s ready
    // ---- phase 2: layer-0 cell/hidden update (owners only)
    if (g < H_) {
      const float iv = g0s[j], fv = g0s[H_ + j], gv = g0s[2*H_ + j], ov = g0s[3*H_ + j];
      c0r = fmaf(fv, c0r, iv * gv);
      h0s[j] = ov * ftanh(c0r);
    }
    __syncthreads();                                        // B2: h0s ready
    // ---- phase 3: layer-1 gate = Wih1*h0_new + Whh1*h1
    float b = bias1;
    #pragma unroll
    for (int i = 0; i < 64; ++i) {
      const uint32_t p = w1[i];
      b = fmaf(bf16lo(p), h0s[2*i],     b);
      b = fmaf(bf16hi(p), h0s[2*i + 1], b);
      const uint32_t q = w2[i];
      b = fmaf(bf16lo(q), h1s[2*i],     b);
      b = fmaf(bf16hi(q), h1s[2*i + 1], b);
    }
    g1s[g] = (gate == 2) ? ftanh(b) : fsigmoid(b);
    __syncthreads();                                        // B3: g1s ready
    // ---- phase 4: layer-1 cell/hidden update (owners only).
    // No trailing barrier: h1s written here is next read in phase 3 of t+1,
    // which is already ordered behind B1(t+1) and B2(t+1); phase 1 of t+1
    // touches only g0s/h0s/X0.
    if (g < H_) {
      const float iv = g1s[j], fv = g1s[H_ + j], gv = g1s[2*H_ + j], ov = g1s[3*H_ + j];
      c1r = fmaf(fv, c1r, iv * gv);
      const float h = ov * ftanh(c1r);
      h1s[j] = h;
      outs[(size_t)t * H_ + j] = f32_to_bf16_rte(h);
    }
  }
}

// --------------------------- head: relu(outs@w1^T+b1)@w2^T+b2 + prev (fused)
__global__ __launch_bounds__(128) void head_kernel(
    const uint16_t* __restrict__ outs,   // [BM,128] bf16
    const uint16_t* __restrict__ w1b,    // [64,128] bf16
    const float* __restrict__ b1,        // [64]
    const float* __restrict__ w2,        // [4,64] f32
    const float* __restrict__ b2,        // [4]
    const float* __restrict__ prev,      // [BM,4]
    float* __restrict__ out) {           // [BM,4]
  __shared__ float hid[16][64];
  const int lane = threadIdx.x & 31;
  const int wave = threadIdx.x >> 5;   // 0..3 -> N tile
  const int half = lane >> 4;
  const int lm   = lane & 15;
  const int m0 = blockIdx.x * 16;
  const int n0 = wave * 16;
  const uint16_t* arow = outs + (size_t)(m0 + lm) * H_;
  const uint16_t* wrow = w1b + (size_t)(n0 + lm) * H_;
  v8f acc = {};
  #pragma unroll
  for (int k0 = 0; k0 < H_; k0 += 32) {
    FragBF fa, fb;
    fa.q[0] = *(const uint4*)(arow + k0 + half * 8);
    fa.q[1] = *(const uint4*)(arow + k0 + 16 + half * 8);
    const uint4* pb = (const uint4*)(wrow + k0 + half * 8);
    fb.q[0] = pb[0];
    fb.q[1] = pb[2];
    acc = __builtin_amdgcn_wmma_f32_16x16x32_bf16(false, fa.v, false, fb.v,
                                                  (short)0, acc, false, false);
  }
  const float bias = b1[n0 + lm];
  #pragma unroll
  for (int r = 0; r < 8; ++r)
    hid[half * 8 + r][n0 + lm] = fmaxf(acc[r] + bias, 0.0f);
  __syncthreads();
  if (threadIdx.x < 64) {
    const int rl = threadIdx.x >> 2;
    const int o  = threadIdx.x & 3;
    float s = b2[o];
    #pragma unroll 8
    for (int k = 0; k < 64; ++k) s = fmaf(hid[rl][k], w2[o * 64 + k], s);
    const size_t row = (size_t)(m0 + rl);
    out[row * 4 + o] = prev[row * 4 + o] + s;
  }
}

// -------------------------------------------------------------------- launch
extern "C" void kernel_launch(void* const* d_in, const int* in_sizes, int n_in,
                              void* d_out, int out_size, void* d_ws, size_t ws_size,
                              hipStream_t stream) {
  (void)in_sizes; (void)n_in; (void)out_size; (void)ws_size;
  const float* mv      = (const float*)d_in[0];
  const float* prev    = (const float*)d_in[1];
  const float* conv1_w = (const float*)d_in[2];
  const float* conv1_b = (const float*)d_in[3];
  const float* conv2_w = (const float*)d_in[4];
  const float* conv2_b = (const float*)d_in[5];
  const float* lin_w   = (const float*)d_in[6];
  const float* lin_b   = (const float*)d_in[7];
  const float* bbox_w  = (const float*)d_in[8];
  const float* bbox_b  = (const float*)d_in[9];
  const float* W_ih0   = (const float*)d_in[10];
  const float* W_hh0   = (const float*)d_in[11];
  const float* b_ih0   = (const float*)d_in[12];
  const float* b_hh0   = (const float*)d_in[13];
  const float* W_ih1   = (const float*)d_in[14];
  const float* W_hh1   = (const float*)d_in[15];
  const float* b_ih1   = (const float*)d_in[16];
  const float* b_hh1   = (const float*)d_in[17];
  const float* pred_w1 = (const float*)d_in[18];
  const float* pred_b1 = (const float*)d_in[19];
  const float* pred_w2 = (const float*)d_in[20];
  const float* pred_b2 = (const float*)d_in[21];
  float* out = (float*)d_out;

  char* ws = (char*)d_ws;
  size_t off = 0;
  auto take = [&](size_t bytes) -> char* {
    char* p = ws + off;
    off = (off + bytes + 255) & ~(size_t)255;
    return p;
  };
  float*    c1out  = (float*)   take((size_t)B_ * 32 * 1600 * 4);   // 26.2 MB
  float*    X0     = (float*)   take((size_t)BM * G4 * 4);          // 26.2 MB
  float*    mfeat  = (float*)   take((size_t)B_ * H_ * 4);
  uint16_t* pooled = (uint16_t*)take((size_t)B_ * 4096 * 2);
  uint16_t* linwb  = (uint16_t*)take((size_t)H_ * 4096 * 2);
  uint16_t* comb   = (uint16_t*)take((size_t)BM * KIN * 2);         // 4.9 MB
  uint16_t* wih0b  = (uint16_t*)take((size_t)G4 * KIN * 2);
  uint16_t* whh0b  = (uint16_t*)take((size_t)G4 * H_ * 2);
  uint16_t* wih1b  = (uint16_t*)take((size_t)G4 * H_ * 2);
  uint16_t* whh1b  = (uint16_t*)take((size_t)G4 * H_ * 2);
  uint16_t* pw1b   = (uint16_t*)take((size_t)64 * H_ * 2);
  uint16_t* outsb  = (uint16_t*)take((size_t)BM * H_ * 2);          // 3.3 MB

  auto cvt = [&](const float* src, uint16_t* dst, int n) {
    cvt_f32_bf16<<<(n + 255) / 256, 256, 0, stream>>>(src, dst, n);
  };
  // weight conversions (bf16 WMMA operand prep)
  cvt(lin_w,   linwb, H_ * 4096);
  cvt(W_ih0,   wih0b, G4 * KIN);
  cvt(W_hh0,   whh0b, G4 * H_);
  cvt(W_ih1,   wih1b, G4 * H_);
  cvt(W_hh1,   whh1b, G4 * H_);
  cvt(pred_w1, pw1b,  64 * H_);

  // CNN front-end
  conv1_kernel<<<(B_ * 32 * 1600 + 255) / 256, 256, 0, stream>>>(mv, conv1_w, conv1_b, c1out);
  conv2_pool_kernel<<<B_ * 8, 256, 0, stream>>>(c1out, conv2_w, conv2_b, pooled);

  // motion_feat[128,128] = pooled @ lin_w^T + lin_b   (WMMA)
  wmma_gemm_xwT<<<dim3(B_ / 16, 1), 256, 0, stream>>>(
      pooled, linwb, lin_b, nullptr, mfeat, B_, H_, 4096);

  // combined[12800,192] bf16
  build_combined_kernel<<<(BM * KIN + 255) / 256, 256, 0, stream>>>(
      mfeat, prev, bbox_w, bbox_b, comb);

  // X0[12800,512] = combined @ W_ih0^T + b_ih0 + b_hh0   (WMMA, hoisted out of scan)
  wmma_gemm_xwT<<<dim3(BM / 16, G4 / 128), 256, 0, stream>>>(
      comb, wih0b, b_ih0, b_hh0, X0, BM, G4, KIN);

  // sequential recurrence (single persistent workgroup)
  lstm_scan_kernel<<<1, 512, 0, stream>>>(X0, whh0b, wih1b, whh1b, b_ih1, b_hh1, outsb);

  // prediction head + residual add (WMMA)
  head_kernel<<<BM / 16, 128, 0, stream>>>(outsb, pw1b, pred_b1, pred_w2, pred_b2, prev, out);
}
